// GNN_NOCAT_52699248722103
// MI455X (gfx1250) — matrix operand order repaired and checked by reading, not compiled, not run on hardware
//
#include <hip/hip_runtime.h>

// ---------------------------------------------------------------------------
// Hetero-GNN forward for MI455X (gfx1250, wave32).
//
// Input layout assumption (d_in): setup_inputs() top-level keys in insertion
// order; the nested 'params' dict is flattened into individual leaves in JAX
// tree-flatten order (dict keys sorted alphabetically, DFS):
//   [0] x_artist  [1] x_track  [2] x_tag  [3] ea_lastfm  [4] ea_follows
//   [5..128] params leaves:
//       W1, W2, b1, b2,
//       l1: collab{Wd,Ws,ad,as,b}, follows{We,Wl,Wr,att,b}, lastfm{We,Wl,Wr,att,b},
//           linked{..}, musical{..}, personal{..},
//           tag_art{Wl,Wp,Wr,bl,bp}, tag_trk{..}, tags_artists{..},
//           tags_tracks{..}, worked_by{..}, worked_in{..},
//       l2: (same key order)
//   [129..140] e_collab, e_lastfm, e_follows, e_linked, e_musical, e_personal,
//              e_tags_artists, e_worked_by, e_tag_art, e_tag_trk,
//              e_tags_tracks, e_worked_in   (each stacked [2,E]: src then dst)
//
// Dense projections run through v_wmma_f32_16x16x32_f16. Tiles are staged in
// LDS *in fragment order* so fragment reads are 2x ds_load_b128 per fragment
// instead of 16x ds_load_u16 gathers. Edge softmax/scatter are wave32-per-edge
// kernels with L2-resident float atomics (dst accumulators fit in 192MB L2).
// Layer-2 tag/track outputs are dead code for the artist output and skipped.
// ---------------------------------------------------------------------------

#define N_A   30000
#define N_T   60000
#define N_G   1000
#define HID   128
#define OUTC  64
#define NHEAD 3
#define HCAT  (NHEAD * HID)   // 384

typedef _Float16 half16  __attribute__((ext_vector_type(16)));
typedef _Float16 half4v  __attribute__((ext_vector_type(4)));
typedef float    floatx8 __attribute__((ext_vector_type(8)));

#define GEMM_ACCUM 1
#define GEMM_RELU  2

// Ordered-uint encoding of float for atomicMax-based segment max.
__device__ __forceinline__ unsigned f32_ord(float f) {
  unsigned u = __float_as_uint(f);
  return (u & 0x80000000u) ? ~u : (u | 0x80000000u);
}
__device__ __forceinline__ float ord_f32(unsigned u) {
  return __uint_as_float((u & 0x80000000u) ? (u & 0x7fffffffu) : ~u);
}
#define ORD_NEG_INF 0x007FFFFFu   // f32_ord(-inf)

// ---------------------------------------------------------------------------
// WMMA GEMM: C[M,N] = A[M,K] @ B[K,N] (+bias) (+=C if ACCUM) (relu if RELU)
// 64x64 C tile per 128-thread block (4 waves; wave w: rows w*16..w*16+15).
// Requires K % 32 == 0 and N % 64 == 0 (true for all call sites: K in
// {64,128,512}, N in {64,128,384,512}).
//
// LDS staging is done in WMMA *fragment order* (ISA 7.12.2):
//   A 16x32 f16 fragment: lane = hh*16 + m, hh = (k>>3)&1,
//                         elem = (k&7) + (k>=16 ? 8 : 0)
//   B 32x16 f16 fragment: lane = hh*16 + n, hh and elem as for A
// so each lane's 16 halfs are contiguous 32B -> 2x ds_load_b128 per fragment.
// ---------------------------------------------------------------------------
__global__ __launch_bounds__(128)
void hg_gemm_wmma(const float* __restrict__ A, const float* __restrict__ B,
                  float* __restrict__ C, const float* __restrict__ bias,
                  int M, int N, int K, int flags)
{
  __shared__ __align__(32) _Float16 AsF[4][32][16];   // [row strip][lane][elem]
  __shared__ __align__(32) _Float16 BsF[4][32][16];   // [col group][lane][elem]

  const int tid  = threadIdx.x;
  const int lane = tid & 31;
  const int wave = tid >> 5;
  const int bm   = blockIdx.y * 64;
  const int bn   = blockIdx.x * 64;

  floatx8 acc[4] = {floatx8{}, floatx8{}, floatx8{}, floatx8{}};

  for (int k0 = 0; k0 < K; k0 += 32) {
    // ---- stage A tile (64 rows x 32 k): float4 global load, b64 LDS store --
#pragma unroll
    for (int i = 0; i < 4; ++i) {
      int t  = tid + i * 128;          // 0..511 tasks
      int r  = t >> 3;                 // 0..63 tile row
      int kq = (t & 7) * 4;            // 0,4,..,28 (4-aligned k quad)
      int gr = bm + r;
      float4 v = make_float4(0.f, 0.f, 0.f, 0.f);
      if (gr < M) v = *(const float4*)(A + (long)gr * K + k0 + kq);
      half4v h;
      h[0] = (_Float16)v.x; h[1] = (_Float16)v.y;
      h[2] = (_Float16)v.z; h[3] = (_Float16)v.w;
      int hh    = (kq >> 3) & 1;
      int ebase = (kq & 7) + ((kq & 16) ? 8 : 0);
      *(half4v*)&AsF[r >> 4][hh * 16 + (r & 15)][ebase] = h;
    }
    // ---- stage B tile (32 k x 64 n): 4 strided loads, b64 LDS store --------
#pragma unroll
    for (int i = 0; i < 4; ++i) {
      int t  = tid + i * 128;
      int c  = t & 63;                 // tile col (coalesced across threads)
      int kq = (t >> 6) * 4;           // 0,4,..,28
      const float* bp = B + (long)(k0 + kq) * N + bn + c;
      half4v h;
#pragma unroll
      for (int j = 0; j < 4; ++j) h[j] = (_Float16)bp[(long)j * N];
      *(half4v*)&BsF[c >> 4][(((kq >> 3) & 1) << 4) + (c & 15)][(kq & 7) + ((kq & 16) ? 8 : 0)] = h;
    }
    __syncthreads();

    half16 af = *(const half16*)&AsF[wave][lane][0];
#pragma unroll
    for (int g = 0; g < 4; ++g) {
      half16 bf = *(const half16*)&BsF[g][lane][0];
      acc[g] = __builtin_amdgcn_wmma_f32_16x16x32_f16(
          false, af, false, bf, (short)0, acc[g], false, false);
    }
    __syncthreads();
  }

  // Store (C/D layout: lane n = lane&15, VGPR r holds M = r + 8*(lane>>4)).
#pragma unroll
  for (int g = 0; g < 4; ++g) {
    int nl = (g << 4) + (lane & 15);
    int gc = bn + nl;
#pragma unroll
    for (int r = 0; r < 8; ++r) {
      int ml = (wave << 4) + r + ((lane >> 4) << 3);
      int gr = bm + ml;
      if (gr < M && gc < N) {
        float v = acc[g][r];
        if (bias) v += bias[gc];
        long off = (long)gr * N + gc;
        if (flags & GEMM_ACCUM) v += C[off];
        if (flags & GEMM_RELU)  v = v > 0.f ? v : 0.f;
        C[off] = v;
      }
    }
  }
}

// ---------------------------------------------------------------------------
// Elementwise / fill / reduce helpers
// ---------------------------------------------------------------------------
__global__ void hg_fill_f32(float* p, float v, long n) {
  long i = blockIdx.x * (long)blockDim.x + threadIdx.x;
  if (i < n) p[i] = v;
}
__global__ void hg_fill_u32(unsigned* p, unsigned v, long n) {
  long i = blockIdx.x * (long)blockDim.x + threadIdx.x;
  if (i < n) p[i] = v;
}
__global__ void hg_reduce_sum(const float* __restrict__ x, long n, float* out) {
  __shared__ float sh[256];
  float a = 0.f;
  for (long i = blockIdx.x * (long)blockDim.x + threadIdx.x; i < n;
       i += (long)gridDim.x * blockDim.x) a += x[i];
  sh[threadIdx.x] = a;
  __syncthreads();
  for (int s = 128; s > 0; s >>= 1) {
    if (threadIdx.x < s) sh[threadIdx.x] += sh[threadIdx.x + s];
    __syncthreads();
  }
  if (threadIdx.x == 0) atomicAdd(out, sh[0]);
}

// a_s[n,h] = dot(xl[n, h*128:...], av[h*128:...]); wave32 per node.
__global__ void hg_rowdot(const float* __restrict__ xl, const float* __restrict__ av,
                          float* __restrict__ out, int n) {
  int node = blockIdx.x * (blockDim.x >> 5) + (threadIdx.x >> 5);
  int lane = threadIdx.x & 31;
  if (node >= n) return;
  const float* r = xl + (long)node * HCAT;
#pragma unroll
  for (int h = 0; h < NHEAD; ++h) {
    float a = 0.f;
    for (int c = lane; c < HID; c += 32) a += r[h * HID + c] * av[h * HID + c];
    for (int o = 16; o > 0; o >>= 1) a += __shfl_xor(a, o, 32);
    if (lane == 0) out[(long)node * NHEAD + h] = a;
  }
}

// ---------------------------------------------------------------------------
// GAT edge kernels (self loops appended: edge e >= E has src=dst=e-E)
// ---------------------------------------------------------------------------
__global__ void hg_gat_alpha(const float* __restrict__ as_, const float* __restrict__ ad_,
                             const int* __restrict__ esrc, const int* __restrict__ edst,
                             int E, int n, float* __restrict__ alpha,
                             unsigned* __restrict__ mmax) {
  long i = blockIdx.x * (long)blockDim.x + threadIdx.x;
  long tot = (long)(E + n) * NHEAD;
  if (i >= tot) return;
  int e = (int)(i / NHEAD), h = (int)(i % NHEAD);
  int s, d;
  if (e < E) { s = esrc[e]; d = edst[e]; } else { s = d = e - E; }
  float v = as_[(long)s * NHEAD + h] + ad_[(long)d * NHEAD + h];
  v = v > 0.f ? v : 0.2f * v;                      // leaky_relu(0.2)
  alpha[i] = v;
  atomicMax(&mmax[(long)d * NHEAD + h], f32_ord(v));
}

// GATv2: wave32 per edge; alpha[e,h] = att_h . leaky(xl[s]+xr[d]+ea*We)
__global__ void hg_gatv2_alpha(const float* __restrict__ xl, const float* __restrict__ xr,
                               const float* __restrict__ We, const float* __restrict__ att,
                               const float* __restrict__ ea, const float* __restrict__ easum,
                               const int* __restrict__ esrc, const int* __restrict__ edst,
                               int E, int n, float* __restrict__ alpha,
                               unsigned* __restrict__ mmax) {
  int e = blockIdx.x * (blockDim.x >> 5) + (threadIdx.x >> 5);
  int lane = threadIdx.x & 31;
  if (e >= E + n) return;
  int s, d; float w;
  if (e < E) { s = esrc[e]; d = edst[e]; w = ea[e]; }
  else       { s = d = e - E; w = easum[0] / (float)E; }
  const float* rl = xl + (long)s * HCAT;
  const float* rr = xr + (long)d * HCAT;
  float dot[NHEAD];
#pragma unroll
  for (int h = 0; h < NHEAD; ++h) {
    float a = 0.f;
    for (int c = lane; c < HID; c += 32) {
      int idx = h * HID + c;
      float v = rl[idx] + rr[idx] + w * We[idx];
      v = v > 0.f ? v : 0.2f * v;
      a += v * att[idx];
    }
    for (int o = 16; o > 0; o >>= 1) a += __shfl_xor(a, o, 32);
    dot[h] = a;
  }
  if (lane == 0) {
#pragma unroll
    for (int h = 0; h < NHEAD; ++h) {
      alpha[(long)e * NHEAD + h] = dot[h];
      atomicMax(&mmax[(long)d * NHEAD + h], f32_ord(dot[h]));
    }
  }
}

__global__ void hg_softmax_exp(float* __restrict__ alpha, const int* __restrict__ edst,
                               int E, int n, const unsigned* __restrict__ mmax,
                               float* __restrict__ ssum) {
  long i = blockIdx.x * (long)blockDim.x + threadIdx.x;
  long tot = (long)(E + n) * NHEAD;
  if (i >= tot) return;
  int e = (int)(i / NHEAD), h = (int)(i % NHEAD);
  int d = (e < E) ? edst[e] : e - E;
  float m = ord_f32(mmax[(long)d * NHEAD + h]);
  float ev = __expf(alpha[i] - m);
  alpha[i] = ev;
  atomicAdd(&ssum[(long)d * NHEAD + h], ev);
}

// out3[d,h,c] += xl[s,h,c] * alpha[e,h] / (ssum[d,h] + 1e-16); wave per edge
__global__ void hg_gat_scatter(const float* __restrict__ xl, const float* __restrict__ alpha,
                               const float* __restrict__ ssum,
                               const int* __restrict__ esrc, const int* __restrict__ edst,
                               int E, int n, float* __restrict__ out3) {
  int e = blockIdx.x * (blockDim.x >> 5) + (threadIdx.x >> 5);
  int lane = threadIdx.x & 31;
  if (e >= E + n) return;
  int s, d;
  if (e < E) { s = esrc[e]; d = edst[e]; } else { s = d = e - E; }
  float coef[NHEAD];
#pragma unroll
  for (int h = 0; h < NHEAD; ++h)
    coef[h] = alpha[(long)e * NHEAD + h] / (ssum[(long)d * NHEAD + h] + 1e-16f);
  const float* src = xl + (long)s * HCAT;
  float* dst = out3 + (long)d * HCAT;
  for (int i = lane; i < HCAT; i += 32)
    atomicAdd(&dst[i], src[i] * coef[i >> 7]);
}

// acc[n,c] += (mean_h out3[n,h,c] + b[c]) * wrel
__global__ void hg_gat_finish(const float* __restrict__ out3, const float* __restrict__ b,
                              float* __restrict__ acc, int n, float wrel) {
  long i = blockIdx.x * (long)blockDim.x + threadIdx.x;
  if (i >= (long)n * HID) return;
  int node = (int)(i >> 7), c = (int)(i & 127);
  const float* r = out3 + (long)node * HCAT;
  float v = (r[c] + r[HID + c] + r[2 * HID + c]) * (1.f / 3.f) + b[c];
  acc[i] += v * wrel;
}

// ---------------------------------------------------------------------------
// SAGE edge kernels
// ---------------------------------------------------------------------------
__global__ void hg_sage_scatter(const float* __restrict__ xp, const int* __restrict__ esrc,
                                const int* __restrict__ edst, int E, int Cs,
                                float* __restrict__ agg, float* __restrict__ cnt) {
  int e = blockIdx.x * (blockDim.x >> 5) + (threadIdx.x >> 5);
  int lane = threadIdx.x & 31;
  if (e >= E) return;
  int s = esrc[e], d = edst[e];
  const float* src = xp + (long)s * Cs;
  float* dst = agg + (long)d * Cs;
  for (int c = lane; c < Cs; c += 32) atomicAdd(&dst[c], src[c]);
  if (lane == 0) atomicAdd(&cnt[d], 1.f);
}
__global__ void hg_sage_div(float* __restrict__ agg, const float* __restrict__ cnt,
                            int nd, int Cs) {
  long i = blockIdx.x * (long)blockDim.x + threadIdx.x;
  if (i >= (long)nd * Cs) return;
  float k = cnt[i / Cs];
  agg[i] /= fmaxf(k, 1.f);
}

// acc[row,:] += normalize(x[row,:]) * wrel  (wave per row)
__global__ void hg_norm_acc(const float* __restrict__ x, float* __restrict__ acc,
                            int n, int C, float wrel) {
  int row = blockIdx.x * (blockDim.x >> 5) + (threadIdx.x >> 5);
  int lane = threadIdx.x & 31;
  if (row >= n) return;
  const float* r = x + (long)row * C;
  float ss = 0.f;
  for (int c = lane; c < C; c += 32) { float v = r[c]; ss += v * v; }
  for (int o = 16; o > 0; o >>= 1) ss += __shfl_xor(ss, o, 32);
  float nrm = fmaxf(sqrtf(ss), 1e-12f);
  float* a = acc + (long)row * C;
  for (int c = lane; c < C; c += 32) a[c] += r[c] / nrm * wrel;
}
__global__ void hg_final_norm(const float* __restrict__ x, float* __restrict__ out,
                              int n, int C) {
  int row = blockIdx.x * (blockDim.x >> 5) + (threadIdx.x >> 5);
  int lane = threadIdx.x & 31;
  if (row >= n) return;
  const float* r = x + (long)row * C;
  float ss = 0.f;
  for (int c = lane; c < C; c += 32) { float v = r[c]; ss += v * v; }
  for (int o = 16; o > 0; o >>= 1) ss += __shfl_xor(ss, o, 32);
  float nrm = fmaxf(sqrtf(ss), 1e-12f);
  float* o2 = out + (long)row * C;
  for (int c = lane; c < C; c += 32) o2[c] = r[c] / nrm;
}

// ---------------------------------------------------------------------------
// Host-side orchestration
// ---------------------------------------------------------------------------
struct GatP  { const float *Wd, *Ws, *ad, *as, *b; };
struct Gv2P  { const float *We, *Wl, *Wr, *att, *b; };
struct SageP { const float *Wl, *Wp, *Wr, *bl, *bp; };
struct LayerP {
  GatP collab; Gv2P follows, lastfm; GatP linked, musical, personal;
  SageP tag_art, tag_trk, tags_artists, tags_tracks, worked_by, worked_in;
};
struct Scratch {
  float *XL, *XR, *OUT3, *AS, *AD, *ALPHA, *SSUM, *CNT, *SCAL;
  unsigned* MMAX;
  float *XP, *AGG, *SOUT;   // aliases of XL / XR / OUT3
};

static inline unsigned hg_blk(long tot, int t) { return (unsigned)((tot + t - 1) / t); }

static void hg_gemm(hipStream_t st, const float* A, const float* B, float* C,
                    int M, int N, int K, const float* bias, int flags) {
  dim3 g((N + 63) / 64, (M + 63) / 64);
  hg_gemm_wmma<<<g, 128, 0, st>>>(A, B, C, bias, M, N, K, flags);
}
static void hg_fillf(hipStream_t st, float* p, float v, long n) {
  hg_fill_f32<<<hg_blk(n, 256), 256, 0, st>>>(p, v, n);
}
static void hg_fillu(hipStream_t st, unsigned* p, unsigned v, long n) {
  hg_fill_u32<<<hg_blk(n, 256), 256, 0, st>>>(p, v, n);
}

static void hg_run_gat(hipStream_t st, Scratch& S, const float* x, int n, int Cin,
                       const int* esrc, const int* edst, int E,
                       const GatP& p, float* acc, float wrel) {
  hg_gemm(st, x, p.Ws, S.XL, n, HCAT, Cin, nullptr, 0);
  hg_gemm(st, x, p.Wd, S.XR, n, HCAT, Cin, nullptr, 0);
  hg_rowdot<<<hg_blk(n, 8), 256, 0, st>>>(S.XL, p.as, S.AS, n);
  hg_rowdot<<<hg_blk(n, 8), 256, 0, st>>>(S.XR, p.ad, S.AD, n);
  hg_fillu(st, S.MMAX, ORD_NEG_INF, (long)n * NHEAD);
  hg_fillf(st, S.SSUM, 0.f, (long)n * NHEAD);
  hg_fillf(st, S.OUT3, 0.f, (long)n * HCAT);
  long tot = (long)(E + n) * NHEAD;
  hg_gat_alpha<<<hg_blk(tot, 256), 256, 0, st>>>(S.AS, S.AD, esrc, edst, E, n, S.ALPHA, S.MMAX);
  hg_softmax_exp<<<hg_blk(tot, 256), 256, 0, st>>>(S.ALPHA, edst, E, n, S.MMAX, S.SSUM);
  hg_gat_scatter<<<hg_blk(E + n, 8), 256, 0, st>>>(S.XL, S.ALPHA, S.SSUM, esrc, edst, E, n, S.OUT3);
  hg_gat_finish<<<hg_blk((long)n * HID, 256), 256, 0, st>>>(S.OUT3, p.b, acc, n, wrel);
}

static void hg_run_gatv2(hipStream_t st, Scratch& S, const float* x, int n, int Cin,
                         const int* esrc, const int* edst, int E, const float* ea,
                         const Gv2P& p, float* acc, float wrel) {
  hg_gemm(st, x, p.Wl, S.XL, n, HCAT, Cin, nullptr, 0);
  hg_gemm(st, x, p.Wr, S.XR, n, HCAT, Cin, nullptr, 0);
  hg_fillf(st, S.SCAL, 0.f, 1);
  unsigned rb = hg_blk(E, 256); if (rb > 1024) rb = 1024;
  hg_reduce_sum<<<rb, 256, 0, st>>>(ea, E, S.SCAL);
  hg_fillu(st, S.MMAX, ORD_NEG_INF, (long)n * NHEAD);
  hg_fillf(st, S.SSUM, 0.f, (long)n * NHEAD);
  hg_fillf(st, S.OUT3, 0.f, (long)n * HCAT);
  hg_gatv2_alpha<<<hg_blk(E + n, 8), 256, 0, st>>>(S.XL, S.XR, p.We, p.att, ea, S.SCAL,
                                                   esrc, edst, E, n, S.ALPHA, S.MMAX);
  long tot = (long)(E + n) * NHEAD;
  hg_softmax_exp<<<hg_blk(tot, 256), 256, 0, st>>>(S.ALPHA, edst, E, n, S.MMAX, S.SSUM);
  hg_gat_scatter<<<hg_blk(E + n, 8), 256, 0, st>>>(S.XL, S.ALPHA, S.SSUM, esrc, edst, E, n, S.OUT3);
  hg_gat_finish<<<hg_blk((long)n * HID, 256), 256, 0, st>>>(S.OUT3, p.b, acc, n, wrel);
}

static void hg_run_sage(hipStream_t st, Scratch& S,
                        const float* xs, int Ns, int Cs,
                        const float* xd, int Nd, int Cd,
                        const int* esrc, const int* edst, int E,
                        const SageP& p, float* acc, float wrel) {
  hg_gemm(st, xs, p.Wp, S.XP, Ns, Cs, Cs, p.bp, GEMM_RELU);        // xp = relu(xs@Wp+bp)
  hg_fillf(st, S.AGG, 0.f, (long)Nd * Cs);
  hg_fillf(st, S.CNT, 0.f, Nd);
  hg_sage_scatter<<<hg_blk(E, 8), 256, 0, st>>>(S.XP, esrc, edst, E, Cs, S.AGG, S.CNT);
  hg_sage_div<<<hg_blk((long)Nd * Cs, 256), 256, 0, st>>>(S.AGG, S.CNT, Nd, Cs);
  hg_gemm(st, S.AGG, p.Wl, S.SOUT, Nd, HID, Cs, p.bl, 0);          // agg@Wl + bl
  hg_gemm(st, xd, p.Wr, S.SOUT, Nd, HID, Cd, nullptr, GEMM_ACCUM); // += xd@Wr
  hg_norm_acc<<<hg_blk(Nd, 8), 256, 0, st>>>(S.SOUT, acc, Nd, HID, wrel);
}

extern "C" void kernel_launch(void* const* d_in, const int* in_sizes, int n_in,
                              void* d_out, int out_size, void* d_ws, size_t ws_size,
                              hipStream_t stream) {
  // ---- inputs ----
  const float* xa0 = (const float*)d_in[0];
  const float* xt0 = (const float*)d_in[1];
  const float* xg0 = (const float*)d_in[2];
  const float* ea_lastfm  = (const float*)d_in[3];
  const float* ea_follows = (const float*)d_in[4];

  int pi = 5;
  auto F   = [&]() { return (const float*)d_in[pi++]; };
  auto gat = [&]() { GatP p;  p.Wd = F(); p.Ws = F(); p.ad = F(); p.as = F(); p.b = F(); return p; };
  auto gv2 = [&]() { Gv2P p;  p.We = F(); p.Wl = F(); p.Wr = F(); p.att = F(); p.b = F(); return p; };
  auto sag = [&]() { SageP p; p.Wl = F(); p.Wp = F(); p.Wr = F(); p.bl = F(); p.bp = F(); return p; };
  const float* W1 = F(); const float* W2 = F();
  const float* b1 = F(); const float* b2 = F();
  auto layer = [&]() {
    LayerP L;
    L.collab = gat(); L.follows = gv2(); L.lastfm = gv2();
    L.linked = gat(); L.musical = gat(); L.personal = gat();
    L.tag_art = sag(); L.tag_trk = sag(); L.tags_artists = sag();
    L.tags_tracks = sag(); L.worked_by = sag(); L.worked_in = sag();
    return L;
  };
  LayerP L1 = layer();
  LayerP L2 = layer();

  int ei = pi;   // 129
  struct Edg { const int *s, *d; int E; };
  auto edg = [&]() { Edg e; e.E = in_sizes[ei] / 2;
                     e.s = (const int*)d_in[ei]; e.d = e.s + e.E; ++ei; return e; };
  Edg e_collab = edg(), e_lastfm = edg(), e_follows = edg(), e_linked = edg(),
      e_musical = edg(), e_personal = edg(), e_tags_artists = edg(),
      e_worked_by = edg(), e_tag_art = edg(), e_tag_trk = edg(),
      e_tags_tracks = edg(), e_worked_in = edg();

  // ---- workspace carve (~220 MB) ----
  char* base = (char*)d_ws; size_t off = 0;
  auto alloc = [&](size_t bytes) -> void* {
    off = (off + 255) & ~(size_t)255; void* p = base + off; off += bytes; return p;
  };
  float* A1 = (float*)alloc((size_t)N_A * HID * 4);
  float* T1 = (float*)alloc((size_t)N_T * HID * 4);
  float* G1 = (float*)alloc((size_t)N_G * HID * 4);
  float* A2 = (float*)alloc((size_t)N_A * HID * 4);
  Scratch S;
  S.XL    = (float*)alloc((size_t)N_A * 512 * 4);       // also MLP hidden
  S.XR    = (float*)alloc((size_t)N_A * HCAT * 4);
  S.OUT3  = (float*)alloc((size_t)N_A * HCAT * 4);
  S.AS    = (float*)alloc((size_t)N_A * NHEAD * 4);
  S.AD    = (float*)alloc((size_t)N_A * NHEAD * 4);
  S.ALPHA = (float*)alloc((size_t)(250000 + N_A) * NHEAD * 4);
  S.MMAX  = (unsigned*)alloc((size_t)N_A * NHEAD * 4);
  S.SSUM  = (float*)alloc((size_t)N_A * NHEAD * 4);
  S.CNT   = (float*)alloc((size_t)N_T * 4);
  S.SCAL  = (float*)alloc(256);
  S.XP = S.XL; S.AGG = S.XR; S.SOUT = S.OUT3;           // sequential reuse

  hipStream_t st = stream;

  // ================= Layer 1 =================
  hg_fillf(st, A1, 0.f, (long)N_A * HID);
  hg_fillf(st, T1, 0.f, (long)N_T * HID);
  hg_fillf(st, G1, 0.f, (long)N_G * HID);

  const float w8 = 1.f / 8.f, w2 = 1.f / 2.f;
  // artist destination (8 relations)
  hg_run_gat  (st, S, xa0, N_A, 128, e_collab.s,   e_collab.d,   e_collab.E,   L1.collab,   A1, w8);
  hg_run_gatv2(st, S, xa0, N_A, 128, e_lastfm.s,   e_lastfm.d,   e_lastfm.E,   ea_lastfm,  L1.lastfm,  A1, w8);
  hg_run_gatv2(st, S, xa0, N_A, 128, e_follows.s,  e_follows.d,  e_follows.E,  ea_follows, L1.follows, A1, w8);
  hg_run_gat  (st, S, xa0, N_A, 128, e_linked.s,   e_linked.d,   e_linked.E,   L1.linked,   A1, w8);
  hg_run_gat  (st, S, xa0, N_A, 128, e_musical.s,  e_musical.d,  e_musical.E,  L1.musical,  A1, w8);
  hg_run_gat  (st, S, xa0, N_A, 128, e_personal.s, e_personal.d, e_personal.E, L1.personal, A1, w8);
  hg_run_sage (st, S, xg0, N_G, 64,  xa0, N_A, 128, e_tags_artists.s, e_tags_artists.d, e_tags_artists.E, L1.tags_artists, A1, w8);
  hg_run_sage (st, S, xt0, N_T, 128, xa0, N_A, 128, e_worked_by.s,    e_worked_by.d,    e_worked_by.E,    L1.worked_by,    A1, w8);
  // tag destination
  hg_run_sage (st, S, xa0, N_A, 128, xg0, N_G, 64, e_tag_art.s, e_tag_art.d, e_tag_art.E, L1.tag_art, G1, w2);
  hg_run_sage (st, S, xt0, N_T, 128, xg0, N_G, 64, e_tag_trk.s, e_tag_trk.d, e_tag_trk.E, L1.tag_trk, G1, w2);
  // track destination
  hg_run_sage (st, S, xg0, N_G, 64,  xt0, N_T, 128, e_tags_tracks.s, e_tags_tracks.d, e_tags_tracks.E, L1.tags_tracks, T1, w2);
  hg_run_sage (st, S, xa0, N_A, 128, xt0, N_T, 128, e_worked_in.s,   e_worked_in.d,   e_worked_in.E,   L1.worked_in,   T1, w2);

  // ================= Layer 2 (artist destination only; tag/track are dead) ==
  hg_fillf(st, A2, 0.f, (long)N_A * HID);
  hg_run_gat  (st, S, A1, N_A, HID, e_collab.s,   e_collab.d,   e_collab.E,   L2.collab,   A2, w8);
  hg_run_gatv2(st, S, A1, N_A, HID, e_lastfm.s,   e_lastfm.d,   e_lastfm.E,   ea_lastfm,  L2.lastfm,  A2, w8);
  hg_run_gatv2(st, S, A1, N_A, HID, e_follows.s,  e_follows.d,  e_follows.E,  ea_follows, L2.follows, A2, w8);
  hg_run_gat  (st, S, A1, N_A, HID, e_linked.s,   e_linked.d,   e_linked.E,   L2.linked,   A2, w8);
  hg_run_gat  (st, S, A1, N_A, HID, e_musical.s,  e_musical.d,  e_musical.E,  L2.musical,  A2, w8);
  hg_run_gat  (st, S, A1, N_A, HID, e_personal.s, e_personal.d, e_personal.E, L2.personal, A2, w8);
  hg_run_sage (st, S, G1, N_G, HID, A1, N_A, HID, e_tags_artists.s, e_tags_artists.d, e_tags_artists.E, L2.tags_artists, A2, w8);
  hg_run_sage (st, S, T1, N_T, HID, A1, N_A, HID, e_worked_by.s,    e_worked_by.d,    e_worked_by.E,    L2.worked_by,    A2, w8);

  // ================= MLP head + final L2 normalize =================
  hg_gemm(st, A2, W1, S.XL, N_A, 4 * HID, HID, b1, GEMM_RELU);   // 30000x512
  hg_gemm(st, S.XL, W2, S.XR, N_A, OUTC, 4 * HID, b2, 0);        // 30000x64
  hg_final_norm<<<hg_blk(N_A, 8), 256, 0, st>>>(S.XR, (float*)d_out, N_A, OUTC);

  (void)n_in; (void)out_size; (void)ws_size; (void)in_sizes;
}
